// MultiHeadAttention_75505525064041
// MI455X (gfx1250) — matrix operand order repaired and checked by reading, not compiled, not run on hardware
//
#include <hip/hip_runtime.h>
#include <hip/hip_bf16.h>

#define SEQ     2048
#define BATCH   2
#define NHEADS  16
#define DK      64
#define DMODEL  1024
#define MROWS   (BATCH * SEQ)          /* 4096 */
#define OUT_ELEMS ((size_t)MROWS * DMODEL)

typedef __attribute__((ext_vector_type(16))) __bf16 v16bf;
typedef __attribute__((ext_vector_type(8)))  __bf16 v8bf;
typedef __attribute__((ext_vector_type(8)))  float  v8f;

__device__ __forceinline__ unsigned short f2bf(float f) {
  unsigned int u = __float_as_uint(f);
  return (unsigned short)((u + 0x7FFFu + ((u >> 16) & 1u)) >> 16);
}

__device__ __forceinline__ v8f wmma_bf16(v16bf a, v16bf b, v8f c) {
  // D = A(16x32 bf16) * B(32x16 bf16) + C(16x16 f32)
  return __builtin_amdgcn_wmma_f32_16x16x32_bf16(false, a, false, b, (short)0, c,
                                                 false, false);
}

// A-fragment: tile layout [rows][32] bf16. Lane holds row M=lane&15;
// halves 0..7 -> K = 8*hi + 0..7, halves 8..15 -> K = 16 + 8*hi + 0..7.
__device__ __forceinline__ v16bf load_frag_a(const unsigned short* tile, int row0, int lane) {
  const int m  = row0 + (lane & 15);
  const int hi = lane >> 4;
  v8bf lo = *(const v8bf*)(tile + m * 32 + 8 * hi);
  v8bf hh = *(const v8bf*)(tile + m * 32 + 16 + 8 * hi);
  return __builtin_shufflevector(lo, hh, 0,1,2,3,4,5,6,7,8,9,10,11,12,13,14,15);
}

// B-fragment: tile staged column-major [cols][32] bf16. Lane holds col N=lane&15;
// halves 0..7 -> K = 8*hi + 0..7, halves 8..15 -> K = 16 + 8*hi + 0..7.
__device__ __forceinline__ v16bf load_frag_b(const unsigned short* tile, int col0, int lane) {
  const int n  = col0 + (lane & 15);
  const int kh = lane >> 4;
  v8bf lo = *(const v8bf*)(tile + n * 32 + 8 * kh);
  v8bf hh = *(const v8bf*)(tile + n * 32 + 16 + 8 * kh);
  return __builtin_shufflevector(lo, hh, 0,1,2,3,4,5,6,7,8,9,10,11,12,13,14,15);
}

// ---------------------------------------------------------------------------
// Generic 4096x1024x1024 GEMM: out = A @ W + bias
// MODE 0: bf16 out, split heads (B,H,S,DK)     (q, v projections)
// MODE 1: bf16 out, split heads T (B,H,DK,S)   (k projection, pre-transposed)
// MODE 2: f32  out, row-major M x N            (final output projection)
// ABF16 : A is bf16 (context) instead of f32.
// Block tile 128(M) x 128(N), 8 waves = 4(M) x 2(N), each wave 32x64.
// ---------------------------------------------------------------------------
template<int MODE, bool ABF16>
__global__ __launch_bounds__(256)
void gemm_kernel(const void* __restrict__ Ap, const float* __restrict__ W,
                 const float* __restrict__ bias, void* __restrict__ outp) {
  __shared__ unsigned short As[128 * 32];
  __shared__ unsigned short Bs[128 * 32];
  const int tid  = threadIdx.x;
  const int lane = tid & 31;
  const int wave = tid >> 5;
  const int wm   = wave & 3;     // 4 along M
  const int wn   = wave >> 2;    // 2 along N
  const int tileM = blockIdx.y * 128;
  const int tileN = blockIdx.x * 128;

  v8f acc[2][4];
  #pragma unroll
  for (int i = 0; i < 2; ++i)
    #pragma unroll
    for (int j = 0; j < 4; ++j)
      #pragma unroll
      for (int e = 0; e < 8; ++e) acc[i][j][e] = 0.0f;

  const float*          Af = (const float*)Ap;
  const unsigned short* Ab = (const unsigned short*)Ap;

  for (int k0 = 0; k0 < DMODEL; k0 += 32) {
    // Stage A tile (128 x 32) into LDS as bf16
    if (!ABF16) {
      #pragma unroll
      for (int i = tid * 4; i < 128 * 32; i += 256 * 4) {
        const int r = i >> 5, c = i & 31;
        const float4 f = *(const float4*)(Af + (size_t)(tileM + r) * DMODEL + k0 + c);
        *(unsigned int*)&As[i]     = (unsigned int)f2bf(f.x) | ((unsigned int)f2bf(f.y) << 16);
        *(unsigned int*)&As[i + 2] = (unsigned int)f2bf(f.z) | ((unsigned int)f2bf(f.w) << 16);
      }
    } else {
      #pragma unroll
      for (int i = tid * 8; i < 128 * 32; i += 256 * 8) {
        const int r = i >> 5, c = i & 31;
        *(uint4*)&As[i] = *(const uint4*)(Ab + (size_t)(tileM + r) * DMODEL + k0 + c);
      }
    }
    // Stage W tile (32 x 128) transposed into LDS: Bs[n][k]
    #pragma unroll
    for (int i = tid * 4; i < 128 * 32; i += 256 * 4) {
      const int n = i & 127, k = i >> 7;
      const float4 f = *(const float4*)(W + (size_t)(k0 + k) * DMODEL + tileN + n);
      Bs[(n + 0) * 32 + k] = f2bf(f.x);
      Bs[(n + 1) * 32 + k] = f2bf(f.y);
      Bs[(n + 2) * 32 + k] = f2bf(f.z);
      Bs[(n + 3) * 32 + k] = f2bf(f.w);
    }
    // Prefetch next k-chunk while WMMAs run (global_prefetch_b8)
    if (k0 + 32 < DMODEL) {
      if (tid < 128) {
        if (!ABF16)
          __builtin_prefetch(Af + (size_t)(tileM + tid) * DMODEL + k0 + 32, 0, 0);
        else
          __builtin_prefetch(Ab + (size_t)(tileM + tid) * DMODEL + k0 + 32, 0, 0);
      } else {
        const int t = tid - 128;
        __builtin_prefetch(W + (size_t)(k0 + 32 + (t & 31)) * DMODEL + tileN + (t >> 5) * 32, 0, 0);
      }
    }
    __syncthreads();

    v16bf a0 = load_frag_a(As, wm * 32 + 0,  lane);
    v16bf a1 = load_frag_a(As, wm * 32 + 16, lane);
    #pragma unroll
    for (int j = 0; j < 4; ++j) {
      v16bf bj = load_frag_b(Bs, wn * 64 + j * 16, lane);
      acc[0][j] = wmma_bf16(a0, bj, acc[0][j]);
      acc[1][j] = wmma_bf16(a1, bj, acc[1][j]);
    }
    __syncthreads();
  }

  const int hsel = lane >> 4;
  const int n15  = lane & 15;
  #pragma unroll
  for (int i = 0; i < 2; ++i)
    #pragma unroll
    for (int j = 0; j < 4; ++j) {
      const int col = tileN + wn * 64 + j * 16 + n15;
      #pragma unroll
      for (int r = 0; r < 8; ++r) {
        const int row = tileM + wm * 32 + i * 16 + r + 8 * hsel;
        const float v = acc[i][j][r] + bias[col];
        if (MODE == 2) {
          ((float*)outp)[(size_t)row * DMODEL + col] = v;
        } else {
          const int b = row >> 11, s = row & (SEQ - 1);
          const int h = col >> 6,  d = col & (DK - 1);
          if (MODE == 0)
            ((unsigned short*)outp)[(((size_t)b * NHEADS + h) * SEQ + s) * DK + d] = f2bf(v);
          else
            ((unsigned short*)outp)[(((size_t)b * NHEADS + h) * DK + d) * SEQ + s] = f2bf(v);
        }
      }
    }
}

// ---------------------------------------------------------------------------
// Scores: per (b,h), S = q(Sx64) @ kT(64xS) * 1/8, masked -> attn (f32, d_out)
// Block tile 128(q) x 128(k), 8 waves = 4(M) x 2(N), each wave 32x64.
// ---------------------------------------------------------------------------
__global__ __launch_bounds__(256)
void scores_kernel(const unsigned short* __restrict__ qbf,
                   const unsigned short* __restrict__ kT,
                   const int* __restrict__ mask,
                   float* __restrict__ attn) {
  __shared__ unsigned short As[128 * 32];
  __shared__ unsigned short Bs[128 * 32];
  const int tid  = threadIdx.x;
  const int lane = tid & 31;
  const int wave = tid >> 5;
  const int wm   = wave & 3;
  const int wn   = wave >> 2;
  const int tileM = blockIdx.y * 128;
  const int tileN = blockIdx.x * 128;
  const int bh = blockIdx.z;
  const int b  = bh >> 4;
  const unsigned short* qb = qbf + (size_t)bh * SEQ * DK;
  const unsigned short* kb = kT  + (size_t)bh * DK * SEQ;

  v8f acc[2][4];
  #pragma unroll
  for (int i = 0; i < 2; ++i)
    #pragma unroll
    for (int j = 0; j < 4; ++j)
      #pragma unroll
      for (int e = 0; e < 8; ++e) acc[i][j][e] = 0.0f;

  #pragma unroll
  for (int k0 = 0; k0 < DK; k0 += 32) {
    // A: q tile 128 x 32 (bf16, row stride DK)
    #pragma unroll
    for (int i = tid * 8; i < 128 * 32; i += 256 * 8) {
      const int r = i >> 5, c = i & 31;
      *(uint4*)&As[i] = *(const uint4*)(qb + (size_t)(tileM + r) * DK + k0 + c);
    }
    // B: kT tile 32 x 128 -> staged column-major Bs[n][k]
    #pragma unroll
    for (int i = tid * 8; i < 128 * 32; i += 256 * 8) {
      const int n0 = i & 127, k = i >> 7;
      uint4 u = *(const uint4*)(kb + (size_t)(k0 + k) * SEQ + tileN + n0);
      const unsigned short* us = (const unsigned short*)&u;
      #pragma unroll
      for (int t = 0; t < 8; ++t) Bs[(n0 + t) * 32 + k] = us[t];
    }
    __syncthreads();

    v16bf a0 = load_frag_a(As, wm * 32 + 0,  lane);
    v16bf a1 = load_frag_a(As, wm * 32 + 16, lane);
    #pragma unroll
    for (int j = 0; j < 4; ++j) {
      v16bf bj = load_frag_b(Bs, wn * 64 + j * 16, lane);
      acc[0][j] = wmma_bf16(a0, bj, acc[0][j]);
      acc[1][j] = wmma_bf16(a1, bj, acc[1][j]);
    }
    __syncthreads();
  }

  const int hsel = lane >> 4;
  const int n15  = lane & 15;
  const float scale = 0.125f;            // 1/sqrt(64)
  #pragma unroll
  for (int i = 0; i < 2; ++i)
    #pragma unroll
    for (int j = 0; j < 4; ++j) {
      const int sk = tileN + wn * 64 + j * 16 + n15;
      const int mz = mask[b * SEQ + sk];
      #pragma unroll
      for (int r = 0; r < 8; ++r) {
        const int sq = tileM + wm * 32 + i * 16 + r + 8 * hsel;
        float v = acc[i][j][r] * scale;
        if (mz == 0) v = -1.0e9f;
        attn[((size_t)bh * SEQ + sq) * SEQ + sk] = v;
      }
    }
}

// ---------------------------------------------------------------------------
// Row softmax over SEQ=2048 floats, in place. One block per row.
// ---------------------------------------------------------------------------
__global__ __launch_bounds__(256)
void softmax_kernel(float* __restrict__ attn) {
  __shared__ float red[256];
  const int tid = threadIdx.x;
  float* p = attn + (size_t)blockIdx.x * SEQ + tid * 8;
  float4 x0 = *(const float4*)(p);
  float4 x1 = *(const float4*)(p + 4);

  float m = fmaxf(fmaxf(fmaxf(x0.x, x0.y), fmaxf(x0.z, x0.w)),
                  fmaxf(fmaxf(x1.x, x1.y), fmaxf(x1.z, x1.w)));
  red[tid] = m;
  __syncthreads();
  for (int s = 128; s > 0; s >>= 1) {
    if (tid < s) red[tid] = fmaxf(red[tid], red[tid + s]);
    __syncthreads();
  }
  m = red[0];
  __syncthreads();

  x0.x = __expf(x0.x - m); x0.y = __expf(x0.y - m);
  x0.z = __expf(x0.z - m); x0.w = __expf(x0.w - m);
  x1.x = __expf(x1.x - m); x1.y = __expf(x1.y - m);
  x1.z = __expf(x1.z - m); x1.w = __expf(x1.w - m);
  red[tid] = (x0.x + x0.y + x0.z + x0.w) + (x1.x + x1.y + x1.z + x1.w);
  __syncthreads();
  for (int s = 128; s > 0; s >>= 1) {
    if (tid < s) red[tid] += red[tid + s];
    __syncthreads();
  }
  const float inv = 1.0f / red[0];

  x0.x *= inv; x0.y *= inv; x0.z *= inv; x0.w *= inv;
  x1.x *= inv; x1.y *= inv; x1.z *= inv; x1.w *= inv;
  *(float4*)(p)     = x0;
  *(float4*)(p + 4) = x1;
}

// ---------------------------------------------------------------------------
// Context: per (b,h), ctx = attn(SxS f32) @ v(Sx64 bf16), K=2048.
// Block tile 256(sq) x 64(d), 8 waves stacked along M, each wave 32x64.
// ---------------------------------------------------------------------------
__global__ __launch_bounds__(256)
void context_kernel(const float* __restrict__ attn,
                    const unsigned short* __restrict__ vbf,
                    unsigned short* __restrict__ ctx) {
  __shared__ unsigned short As[256 * 32];
  __shared__ unsigned short Bs[64 * 32];
  const int tid  = threadIdx.x;
  const int lane = tid & 31;
  const int wave = tid >> 5;            // wm = wave, 8 along M
  const int tileM = blockIdx.y * 256;
  const int bh = blockIdx.z;
  const int b  = bh >> 4;
  const int h  = bh & 15;
  const float*          ab = attn + (size_t)bh * SEQ * SEQ;
  const unsigned short* vb = vbf  + (size_t)bh * SEQ * DK;

  v8f acc[2][4];
  #pragma unroll
  for (int i = 0; i < 2; ++i)
    #pragma unroll
    for (int j = 0; j < 4; ++j)
      #pragma unroll
      for (int e = 0; e < 8; ++e) acc[i][j][e] = 0.0f;

  for (int k0 = 0; k0 < SEQ; k0 += 32) {
    // A: attention tile 256 x 32, f32 -> bf16
    #pragma unroll
    for (int i = tid * 4; i < 256 * 32; i += 256 * 4) {
      const int r = i >> 5, c = i & 31;
      const float4 f = *(const float4*)(ab + (size_t)(tileM + r) * SEQ + k0 + c);
      *(unsigned int*)&As[i]     = (unsigned int)f2bf(f.x) | ((unsigned int)f2bf(f.y) << 16);
      *(unsigned int*)&As[i + 2] = (unsigned int)f2bf(f.z) | ((unsigned int)f2bf(f.w) << 16);
    }
    // B: v tile 32 x 64 (bf16) -> staged column-major Bs[n][k]
    {
      const int i = tid * 8;            // 64*32 = 2048 = 256*8, one pass
      const int n0 = i & 63, k = i >> 6;
      uint4 u = *(const uint4*)(vb + (size_t)(k0 + k) * DK + n0);
      const unsigned short* us = (const unsigned short*)&u;
      #pragma unroll
      for (int t = 0; t < 8; ++t) Bs[(n0 + t) * 32 + k] = us[t];
    }
    // Prefetch next attention k-chunk (one row per thread)
    if (k0 + 32 < SEQ)
      __builtin_prefetch(ab + (size_t)(tileM + tid) * SEQ + k0 + 32, 0, 0);
    __syncthreads();

    v16bf a0 = load_frag_a(As, wave * 32 + 0,  lane);
    v16bf a1 = load_frag_a(As, wave * 32 + 16, lane);
    #pragma unroll
    for (int j = 0; j < 4; ++j) {
      v16bf bj = load_frag_b(Bs, j * 16, lane);
      acc[0][j] = wmma_bf16(a0, bj, acc[0][j]);
      acc[1][j] = wmma_bf16(a1, bj, acc[1][j]);
    }
    __syncthreads();
  }

  const int hsel = lane >> 4;
  const int n15  = lane & 15;
  #pragma unroll
  for (int i = 0; i < 2; ++i)
    #pragma unroll
    for (int j = 0; j < 4; ++j) {
      const int d = j * 16 + n15;
      #pragma unroll
      for (int r = 0; r < 8; ++r) {
        const int sq = tileM + wave * 32 + i * 16 + r + 8 * hsel;
        ctx[((size_t)(b * SEQ + sq)) * DMODEL + h * DK + d] = f2bf(acc[i][j][r]);
      }
    }
}

// ---------------------------------------------------------------------------
extern "C" void kernel_launch(void* const* d_in, const int* in_sizes, int n_in,
                              void* d_out, int out_size, void* d_ws, size_t ws_size,
                              hipStream_t stream) {
  const float* Q    = (const float*)d_in[0];
  const float* K    = (const float*)d_in[1];
  const float* V    = (const float*)d_in[2];
  const int*   mask = (const int*)  d_in[3];
  const float* WQ = (const float*)d_in[4];  const float* bQ = (const float*)d_in[5];
  const float* WK = (const float*)d_in[6];  const float* bK = (const float*)d_in[7];
  const float* WV = (const float*)d_in[8];  const float* bV = (const float*)d_in[9];
  const float* WO = (const float*)d_in[10]; const float* bO = (const float*)d_in[11];

  float* out  = (float*)d_out;
  float* attn = out + OUT_ELEMS;          // attention probs live in d_out

  const size_t perTensor = (size_t)BATCH * NHEADS * SEQ * DK;   // bf16 elems
  unsigned short* qbf = (unsigned short*)d_ws;
  unsigned short* kT  = qbf + perTensor;
  unsigned short* vbf = kT  + perTensor;
  unsigned short* ctx = vbf + perTensor;

  dim3 blk(256);
  dim3 gProj(DMODEL / 128, MROWS / 128);                // (8, 32)
  gemm_kernel<0, false><<<gProj, blk, 0, stream>>>(Q, WQ, bQ, qbf);
  gemm_kernel<1, false><<<gProj, blk, 0, stream>>>(K, WK, bK, kT);
  gemm_kernel<0, false><<<gProj, blk, 0, stream>>>(V, WV, bV, vbf);

  dim3 gSc(SEQ / 128, SEQ / 128, BATCH * NHEADS);       // (16, 16, 32)
  scores_kernel<<<gSc, blk, 0, stream>>>(qbf, kT, mask, attn);

  softmax_kernel<<<dim3(BATCH * NHEADS * SEQ), blk, 0, stream>>>(attn);

  dim3 gCtx(1, SEQ / 256, BATCH * NHEADS);              // (1, 8, 32)
  context_kernel<<<gCtx, blk, 0, stream>>>(attn, vbf, ctx);

  gemm_kernel<2, true><<<gProj, blk, 0, stream>>>(ctx, WO, bO, out);
}
